// Attention_42142219108972
// MI455X (gfx1250) — matrix operand order repaired and checked by reading, not compiled, not run on hardware
//
#include <hip/hip_runtime.h>
#include <hip/hip_bf16.h>

#define DEVI __device__ __forceinline__

typedef __attribute__((ext_vector_type(16))) __bf16 v16bf;
typedef __attribute__((ext_vector_type(8)))  float  v8f;

static constexpr int Bn = 16, S = 1024, D = 512, H = 8, DH = 64;
static constexpr int ROWS = Bn * S;            // 16384

union Frag { v16bf v; unsigned u[8]; };

DEVI unsigned short f2bf(float x) {
  unsigned u = __float_as_uint(x);
  unsigned r = u + 0x7FFFu + ((u >> 16) & 1u);   // round-to-nearest-even
  return (unsigned short)(r >> 16);
}

DEVI v8f wmma_bf16(v16bf a, v16bf b, v8f c) {
  return __builtin_amdgcn_wmma_f32_16x16x32_bf16(false, a, false, b, (short)0, c,
                                                 false, false);
}

// A fragment: 16x32 bf16, row-major source, leading dim `ld` (elements).
// Lane L<16: row=L, VGPR j<4 -> K={2j,2j+1}, j>=4 -> K={16+2(j-4),..}; lanes 16-31: +8.
DEVI v16bf load_a(const unsigned short* __restrict__ src, int ld) {
  int lane = threadIdx.x & 31;
  int row = lane & 15, half = lane >> 4;
  const unsigned short* r = src + row * ld;
  Frag f;
#pragma unroll
  for (int j = 0; j < 8; ++j) {
    int k = (j < 4) ? (2 * j + half * 8) : (16 + 2 * (j - 4) + half * 8);
    f.u[j] = *(const unsigned*)(r + k);          // two packed bf16, 4B aligned
  }
  return f.v;
}

// A fragment from LDS buffer laid out as 16 rows x 32 cols bf16.
DEVI v16bf load_a_lds(const unsigned short* lds) {
  int lane = threadIdx.x & 31;
  int row = lane & 15, half = lane >> 4;
  const unsigned short* r = lds + row * 32;
  Frag f;
#pragma unroll
  for (int j = 0; j < 8; ++j) {
    int k = (j < 4) ? (2 * j + half * 8) : (16 + 2 * (j - 4) + half * 8);
    f.u[j] = *(const unsigned*)(r + k);
  }
  return f.v;
}

// B fragment: 32x16 (KxN) bf16 where element (k,n) lives at src[n*ld + k]
// (source stored N-major so K-pairs are contiguous).
// Lane L<16: col=L, VGPR j<4 -> K={2j,2j+1}, j>=4 -> K={16+2(j-4),..}; lanes 16-31: +8.
DEVI v16bf load_bt(const unsigned short* __restrict__ src, int ld) {
  int lane = threadIdx.x & 31;
  int col = lane & 15, half = lane >> 4;
  const unsigned short* c = src + col * ld;
  Frag f;
#pragma unroll
  for (int j = 0; j < 8; ++j)
    f.u[j] = *(const unsigned*)(c + ((j < 4) ? (2 * j + half * 8) : (16 + 2 * (j - 4) + half * 8)));
  return f.v;
}

DEVI float rmax16(float v) {
#pragma unroll
  for (int m = 1; m < 16; m <<= 1) v = fmaxf(v, __shfl_xor(v, m, 32));
  return v;
}
DEVI float rsum16(float v) {
#pragma unroll
  for (int m = 1; m < 16; m <<= 1) v += __shfl_xor(v, m, 32);
  return v;
}

// ---------------- K0a: f32 -> bf16, 2 elements/thread ----------------
__global__ void cvt_pairs(const float* __restrict__ in,
                          unsigned short* __restrict__ out) {
  int i = blockIdx.x * blockDim.x + threadIdx.x;           // pair index
  float2 f = ((const float2*)in)[i];
  unsigned lo = f2bf(f.x), hi = f2bf(f.y);
  ((unsigned*)out)[i] = lo | (hi << 16);
}

// ---------------- K0b: f32 W[k][n] -> bf16 W^T[n][k] ----------------
__global__ void cvt_w_transpose(const float* __restrict__ w,
                                unsigned short* __restrict__ wt) {
  int idx = blockIdx.x * blockDim.x + threadIdx.x;         // n*512 + k
  int n = idx >> 9, k = idx & 511;
  wt[idx] = f2bf(w[k * D + n]);
}

// ---------------- K1: fused QKV projection ----------------
// wave -> (4 M-tiles, 1 N-tile) x {Q,K,V}: 3 B fragments amortized over 12 WMMAs.
// launch_bounds(256,1): full VGPR budget so the 12 f32 accumulators never spill.
__global__ void __launch_bounds__(256, 1)
qkv_gemm(const unsigned short* __restrict__ xb,
         const unsigned short* __restrict__ wqt,
         const unsigned short* __restrict__ wkt,
         const unsigned short* __restrict__ wvt,
         unsigned short* __restrict__ qb,
         unsigned short* __restrict__ kb,
         unsigned short* __restrict__ vtb) {
  int wave = (blockIdx.x * blockDim.x + threadIdx.x) >> 5;
  int mblk = wave >> 5;                  // 64 rows per wave
  int ntile = wave & 31;
  int lane = threadIdx.x & 31, half = lane >> 4, lm = lane & 15;

  v8f aq[4] = {}, ak[4] = {}, av[4] = {};
  const unsigned short* xrow = xb + mblk * 64 * D;
  const unsigned short* wq = wqt + ntile * 16 * D;
  const unsigned short* wk = wkt + ntile * 16 * D;
  const unsigned short* wv = wvt + ntile * 16 * D;
  for (int ks = 0; ks < D; ks += 32) {
    v16bf A0 = load_a(xrow + 0 * 16 * D + ks, D);
    v16bf A1 = load_a(xrow + 1 * 16 * D + ks, D);
    v16bf A2 = load_a(xrow + 2 * 16 * D + ks, D);
    v16bf A3 = load_a(xrow + 3 * 16 * D + ks, D);
    v16bf BQ = load_bt(wq + ks, D);
    v16bf BK = load_bt(wk + ks, D);
    v16bf BV = load_bt(wv + ks, D);
    aq[0] = wmma_bf16(A0, BQ, aq[0]);
    aq[1] = wmma_bf16(A1, BQ, aq[1]);
    aq[2] = wmma_bf16(A2, BQ, aq[2]);
    aq[3] = wmma_bf16(A3, BQ, aq[3]);
    ak[0] = wmma_bf16(A0, BK, ak[0]);
    ak[1] = wmma_bf16(A1, BK, ak[1]);
    ak[2] = wmma_bf16(A2, BK, ak[2]);
    ak[3] = wmma_bf16(A3, BK, ak[3]);
    av[0] = wmma_bf16(A0, BV, av[0]);
    av[1] = wmma_bf16(A1, BV, av[1]);
    av[2] = wmma_bf16(A2, BV, av[2]);
    av[3] = wmma_bf16(A3, BV, av[3]);
  }
  int col = ntile * 16 + lm;
  int h = col >> 6, dh = col & 63;
#pragma unroll
  for (int mi = 0; mi < 4; ++mi) {
#pragma unroll
    for (int i = 0; i < 8; ++i) {
      int row = mblk * 64 + mi * 16 + i + half * 8;
      qb[row * D + col] = f2bf(aq[mi][i] * 0.125f);   // scale baked into Q
      kb[row * D + col] = f2bf(ak[mi][i]);
      int b = row >> 10, s = row & 1023;
      vtb[((b * H + h) * DH + dh) * S + s] = f2bf(av[mi][i]);
    }
  }
}

// ---------------- K2: ragged flash attention ----------------
// wave -> (b, h, 16-query tile). KV blocks of 32, online softmax f32.
__global__ void attention(const unsigned short* __restrict__ qb,
                          const unsigned short* __restrict__ kb,
                          const unsigned short* __restrict__ vtb,
                          const int* __restrict__ seq_lens,
                          unsigned short* __restrict__ ob) {
  __shared__ unsigned short pbuf[8][16 * 32];   // per-wave P bounce (8 KB)
  int wg = (blockIdx.x * blockDim.x + threadIdx.x) >> 5;
  int wl = threadIdx.x >> 5;
  int qt = wg & 63;
  int h = (wg >> 6) & 7;
  int b = wg >> 9;
  int lane = threadIdx.x & 31, half = lane >> 4, lm = lane & 15;

  int len = seq_lens[b];
  int nkv = (len + 31) >> 5;

  const unsigned short* qbase = qb + (b * S + qt * 16) * D + h * DH;
  v16bf Aq0 = load_a(qbase, D);        // dh 0..31
  v16bf Aq1 = load_a(qbase + 32, D);   // dh 32..63

  v8f o0 = {}, o1 = {}, o2 = {}, o3 = {};
  float m[8], l[8];
#pragma unroll
  for (int i = 0; i < 8; ++i) { m[i] = -3.0e38f; l[i] = 0.0f; }

  for (int kv = 0; kv < nkv; ++kv) {
    int kvb = kv * 32;
    const unsigned short* kbase = kb + (b * S + kvb) * D + h * DH;
    // scores S[16q x 32k] = Q(16x64) @ K^T(64x32): two C sub-tiles
    v16bf Bk00 = load_bt(kbase, D);
    v16bf Bk01 = load_bt(kbase + 32, D);
    v16bf Bk10 = load_bt(kbase + 16 * D, D);
    v16bf Bk11 = load_bt(kbase + 16 * D + 32, D);
    v8f c0 = {}, c1 = {};
    c0 = wmma_bf16(Aq0, Bk00, c0);
    c0 = wmma_bf16(Aq1, Bk01, c0);
    c1 = wmma_bf16(Aq0, Bk10, c1);
    c1 = wmma_bf16(Aq1, Bk11, c1);

    int col0 = kvb + lm;
    bool v0 = col0 < len, v1 = (col0 + 16) < len;
#pragma unroll
    for (int i = 0; i < 8; ++i) {
      float s0 = v0 ? c0[i] : -3.0e38f;
      float s1 = v1 ? c1[i] : -3.0e38f;
      float bm = rmax16(fmaxf(s0, s1));
      float mn = fmaxf(m[i], bm);
      float sc = __expf(m[i] - mn);
      float p0 = __expf(s0 - mn);
      float p1 = __expf(s1 - mn);
      l[i] = l[i] * sc + rsum16(p0 + p1);
      m[i] = mn;
      o0[i] *= sc; o1[i] *= sc; o2[i] *= sc; o3[i] *= sc;
      int r = (i + half * 8) * 32;
      pbuf[wl][r + lm]      = f2bf(p0);   // C-layout -> LDS
      pbuf[wl][r + 16 + lm] = f2bf(p1);
    }
    v16bf P = load_a_lds(&pbuf[wl][0]);   // LDS -> A-layout (ds_load)
    const unsigned short* vb0 = vtb + ((b * H + h) * DH) * S + kvb;
    v16bf Bv0 = load_bt(vb0 +  0 * S, S);
    v16bf Bv1 = load_bt(vb0 + 16 * S, S);
    v16bf Bv2 = load_bt(vb0 + 32 * S, S);
    v16bf Bv3 = load_bt(vb0 + 48 * S, S);
    o0 = wmma_bf16(P, Bv0, o0);
    o1 = wmma_bf16(P, Bv1, o1);
    o2 = wmma_bf16(P, Bv2, o2);
    o3 = wmma_bf16(P, Bv3, o3);
  }

#pragma unroll
  for (int i = 0; i < 8; ++i) {
    int row = b * S + qt * 16 + i + half * 8;
    float inv = 1.0f / l[i];
    int cb = h * DH + lm;
    ob[row * D + cb +  0] = f2bf(o0[i] * inv);
    ob[row * D + cb + 16] = f2bf(o1[i] * inv);
    ob[row * D + cb + 32] = f2bf(o2[i] * inv);
    ob[row * D + cb + 48] = f2bf(o3[i] * inv);
  }
}

// ---------------- K3: output projection + bias + padded-row zeroing ----------------
// wave -> (4 M-tiles, 2 N-tiles): 8 WMMAs per k-step over 6 fragment loads.
__global__ void __launch_bounds__(256, 1)
out_proj(const unsigned short* __restrict__ ob,
         const unsigned short* __restrict__ wot,
         const float* __restrict__ bo,
         const int* __restrict__ seq_lens,
         float* __restrict__ out) {
  int wave = (blockIdx.x * blockDim.x + threadIdx.x) >> 5;
  int mblk = wave >> 4;                  // 64 rows per wave
  int npair = wave & 15;                 // 2 N-tiles per wave
  int lane = threadIdx.x & 31, half = lane >> 4, lm = lane & 15;

  v8f acc[4][2] = {};
  const unsigned short* arow = ob + mblk * 64 * D;
  const unsigned short* bc0 = wot + (npair * 2 + 0) * 16 * D;
  const unsigned short* bc1 = wot + (npair * 2 + 1) * 16 * D;
  for (int ks = 0; ks < D; ks += 32) {
    v16bf A0 = load_a(arow + 0 * 16 * D + ks, D);
    v16bf A1 = load_a(arow + 1 * 16 * D + ks, D);
    v16bf A2 = load_a(arow + 2 * 16 * D + ks, D);
    v16bf A3 = load_a(arow + 3 * 16 * D + ks, D);
    v16bf B0 = load_bt(bc0 + ks, D);
    v16bf B1 = load_bt(bc1 + ks, D);
    acc[0][0] = wmma_bf16(A0, B0, acc[0][0]);
    acc[1][0] = wmma_bf16(A1, B0, acc[1][0]);
    acc[2][0] = wmma_bf16(A2, B0, acc[2][0]);
    acc[3][0] = wmma_bf16(A3, B0, acc[3][0]);
    acc[0][1] = wmma_bf16(A0, B1, acc[0][1]);
    acc[1][1] = wmma_bf16(A1, B1, acc[1][1]);
    acc[2][1] = wmma_bf16(A2, B1, acc[2][1]);
    acc[3][1] = wmma_bf16(A3, B1, acc[3][1]);
  }
  int b = (mblk * 64) >> 10;             // 64-row block never crosses a batch
  int len = seq_lens[b];
#pragma unroll
  for (int nj = 0; nj < 2; ++nj) {
    int col = (npair * 2 + nj) * 16 + lm;
    float bias = bo[col];
#pragma unroll
    for (int mi = 0; mi < 4; ++mi) {
#pragma unroll
      for (int i = 0; i < 8; ++i) {
        int row = mblk * 64 + mi * 16 + i + half * 8;
        int s = row & 1023;
        out[row * D + col] = (s < len) ? (acc[mi][nj][i] + bias) : 0.0f;
      }
    }
  }
}

extern "C" void kernel_launch(void* const* d_in, const int* in_sizes, int n_in,
                              void* d_out, int out_size, void* d_ws, size_t ws_size,
                              hipStream_t stream) {
  const float* x        = (const float*)d_in[0];
  const int*   seq_lens = (const int*)d_in[1];
  const float* Wq       = (const float*)d_in[2];
  const float* Wk       = (const float*)d_in[3];
  const float* Wv       = (const float*)d_in[4];
  const float* Wo       = (const float*)d_in[5];
  const float* bo       = (const float*)d_in[6];
  float* out = (float*)d_out;

  char* ws = (char*)d_ws;
  size_t off = 0;
  auto alloc = [&](size_t bytes) { char* p = ws + off; off += bytes; return p; };
  unsigned short* xb  = (unsigned short*)alloc((size_t)ROWS * D * 2);  // 16 MB
  unsigned short* wqt = (unsigned short*)alloc((size_t)D * D * 2);
  unsigned short* wkt = (unsigned short*)alloc((size_t)D * D * 2);
  unsigned short* wvt = (unsigned short*)alloc((size_t)D * D * 2);
  unsigned short* wot = (unsigned short*)alloc((size_t)D * D * 2);
  unsigned short* qb  = (unsigned short*)alloc((size_t)ROWS * D * 2);
  unsigned short* kb  = (unsigned short*)alloc((size_t)ROWS * D * 2);
  unsigned short* vtb = (unsigned short*)alloc((size_t)ROWS * D * 2);
  unsigned short* ob  = (unsigned short*)alloc((size_t)ROWS * D * 2);
  (void)ws_size; (void)in_sizes; (void)n_in; (void)out_size;

  // K0: conversions
  cvt_pairs<<<(ROWS * D / 2) / 256, 256, 0, stream>>>(x, xb);
  cvt_w_transpose<<<(D * D) / 256, 256, 0, stream>>>(Wq, wqt);
  cvt_w_transpose<<<(D * D) / 256, 256, 0, stream>>>(Wk, wkt);
  cvt_w_transpose<<<(D * D) / 256, 256, 0, stream>>>(Wv, wvt);
  cvt_w_transpose<<<(D * D) / 256, 256, 0, stream>>>(Wo, wot);

  // K1: QKV projection — 256 mblks * 32 ntiles = 8192 waves (8 waves/block)
  qkv_gemm<<<(ROWS / 64) * 32 / 8, 256, 0, stream>>>(xb, wqt, wkt, wvt, qb, kb, vtb);

  // K2: attention — 16*8*64 = 8192 waves
  attention<<<Bn * H * (S / 16) / 8, 256, 0, stream>>>(qb, kb, vtb, seq_lens, ob);

  // K3: output projection — 256 mblks * 16 npairs = 4096 waves
  out_proj<<<(ROWS / 64) * 16 / 8, 256, 0, stream>>>(ob, wot, bo, seq_lens, out);
}
